// Decoder_19877108646609
// MI455X (gfx1250) — matrix operand order repaired and checked by reading, not compiled, not run on hardware
//
#include <hip/hip_runtime.h>
#include <hip/hip_bf16.h>
#include <math.h>

// ---------------- constants (from reference shapes) ----------------
#define TT 512
#define BB 64
#define HH 128
#define VV 32000
#define EE 128
#define MAXLEN 30
#define K2 256              // 2*H = E+H = GEMM K
#define SOS 1

typedef __attribute__((ext_vector_type(16))) __bf16 v16bf;
typedef __attribute__((ext_vector_type(8)))  float  v8f;

static __device__ __forceinline__ __bf16 f2bf(float f) {
    unsigned u; __builtin_memcpy(&u, &f, 4);
    unsigned short r = (unsigned short)((u + 0x7FFFu + ((u >> 16) & 1u)) >> 16);
    __bf16 b; __builtin_memcpy(&b, &r, 2);
    return b;
}

static __device__ __forceinline__ float sigmoidf(float x) {
    return 1.0f / (1.0f + expf(-x));
}

// ---------------- one-time kernels ----------------
__global__ void cvt_wp_bf16(const float* __restrict__ wp, __bf16* __restrict__ wpb, int n) {
    for (int i = blockIdx.x * blockDim.x + threadIdx.x; i < n; i += gridDim.x * blockDim.x)
        wpb[i] = f2bf(wp[i]);
}

__global__ void zero_f32(float* __restrict__ p, int n) {
    int i = blockIdx.x * blockDim.x + threadIdx.x;
    if (i < n) p[i] = 0.0f;
}

// ---------------- per-step: argmax of previous pred + embedding gather ----------------
__global__ void argmax_gather(const float* __restrict__ out, const float* __restrict__ emb,
                              float* __restrict__ embed, int t) {
    int b = blockIdx.x;
    __shared__ float smax[256];
    __shared__ int   sidx[256];
    __shared__ int   widx;
    if (t == 0) {
        if (threadIdx.x == 0) widx = SOS;
    } else {
        const float* row = out + (size_t)b * MAXLEN * VV + (size_t)(t - 1) * VV;
        float best = -INFINITY; int bi = 0;
        for (int v = threadIdx.x; v < VV; v += 256) {
            float x = row[v];
            if (x > best) { best = x; bi = v; }
        }
        smax[threadIdx.x] = best; sidx[threadIdx.x] = bi;
        __syncthreads();
        for (int s = 128; s > 0; s >>= 1) {
            if (threadIdx.x < s) {
                float o = smax[threadIdx.x + s]; int oi = sidx[threadIdx.x + s];
                if (o > smax[threadIdx.x] ||
                    (o == smax[threadIdx.x] && oi < sidx[threadIdx.x])) {
                    smax[threadIdx.x] = o; sidx[threadIdx.x] = oi;
                }
            }
            __syncthreads();
        }
        if (threadIdx.x == 0) widx = sidx[0];
    }
    __syncthreads();
    int idx = widx;
    if (threadIdx.x < HH)
        embed[b * HH + threadIdx.x] = emb[(size_t)idx * EE + threadIdx.x];
}

// ---------------- per-step: LSTM cell 1 (input = [embed, ctx], 256 wide) ----------------
__global__ void lstm_cell1(const float* __restrict__ embed, const float* __restrict__ ctx,
                           const float* __restrict__ Wih, const float* __restrict__ Whh,
                           const float* __restrict__ bih, const float* __restrict__ bhh,
                           float* __restrict__ h, float* __restrict__ c) {
    int tid = threadIdx.x;
    int lb = tid >> 7;                 // 0..1 local batch
    int j  = tid & 127;                // hidden index
    int b  = blockIdx.x * 2 + lb;
    __shared__ float s_in[2][K2];
    __shared__ float s_h[2][HH];
    s_in[lb][j]      = embed[b * HH + j];
    s_in[lb][HH + j] = ctx[b * HH + j];
    s_h[lb][j]       = h[b * HH + j];
    __syncthreads();
    float acc[4];
#pragma unroll
    for (int g = 0; g < 4; ++g) {
        int row = g * HH + j;
        const float* wi = Wih + (size_t)row * K2;
        const float* wh = Whh + (size_t)row * HH;
        float a = bih[row] + bhh[row];
        for (int k = 0; k < K2; ++k) a += s_in[lb][k] * wi[k];
        for (int k = 0; k < HH; ++k) a += s_h[lb][k] * wh[k];
        acc[g] = a;
    }
    float cv = c[b * HH + j];
    float cn = sigmoidf(acc[1]) * cv + sigmoidf(acc[0]) * tanhf(acc[2]);
    float hn = sigmoidf(acc[3]) * tanhf(cn);
    c[b * HH + j] = cn;
    h[b * HH + j] = hn;
}

// ---------------- per-step: LSTM cell 2 (input = h1, 128 wide) ----------------
__global__ void lstm_cell2(const float* __restrict__ h1,
                           const float* __restrict__ Wih, const float* __restrict__ Whh,
                           const float* __restrict__ bih, const float* __restrict__ bhh,
                           float* __restrict__ h, float* __restrict__ c) {
    int tid = threadIdx.x;
    int lb = tid >> 7;
    int j  = tid & 127;
    int b  = blockIdx.x * 2 + lb;
    __shared__ float s_in[2][HH];
    __shared__ float s_h[2][HH];
    s_in[lb][j] = h1[b * HH + j];
    s_h[lb][j]  = h[b * HH + j];
    __syncthreads();
    float acc[4];
#pragma unroll
    for (int g = 0; g < 4; ++g) {
        int row = g * HH + j;
        const float* wi = Wih + (size_t)row * HH;
        const float* wh = Whh + (size_t)row * HH;
        float a = bih[row] + bhh[row];
        for (int k = 0; k < HH; ++k) a += s_in[lb][k] * wi[k] + s_h[lb][k] * wh[k];
        acc[g] = a;
    }
    float cv = c[b * HH + j];
    float cn = sigmoidf(acc[1]) * cv + sigmoidf(acc[0]) * tanhf(acc[2]);
    float hn = sigmoidf(acc[3]) * tanhf(cn);
    c[b * HH + j] = cn;
    h[b * HH + j] = hn;
}

// ---------------- per-step: q -> energy -> masked softmax -> ctx; writes bf16 A=[h2|ctx] ----------------
__global__ void attention(const float* __restrict__ h2, const float* __restrict__ key,
                          const float* __restrict__ values, const float* __restrict__ mask,
                          const float* __restrict__ Wq, const float* __restrict__ bq,
                          float* __restrict__ ctx, __bf16* __restrict__ A_bf) {
    int b = blockIdx.x;
    int tid = threadIdx.x;
    __shared__ float sh2[HH];
    __shared__ float sq[HH];
    __shared__ float sm[TT];     // exp-weights then normalized m
    __shared__ float red[256];

    if (tid < HH) sh2[tid] = h2[b * HH + tid];
    __syncthreads();

    // q = h2 @ Wq.T + bq
    if (tid < HH) {
        const float* w = Wq + (size_t)tid * HH;
        float a = bq[tid];
        for (int k = 0; k < HH; ++k) a += sh2[k] * w[k];
        sq[tid] = a;
    }
    __syncthreads();

    // energy for tt = tid and tid+256
    float e0 = 0.f, e1 = 0.f;
    {
        const float* k0 = key + ((size_t)tid * BB + b) * HH;
        const float* k1 = key + ((size_t)(tid + 256) * BB + b) * HH;
        for (int h = 0; h < HH; ++h) { e0 += sq[h] * k0[h]; e1 += sq[h] * k1[h]; }
    }
    // max reduce
    red[tid] = fmaxf(e0, e1);
    __syncthreads();
    for (int s = 128; s > 0; s >>= 1) {
        if (tid < s) red[tid] = fmaxf(red[tid], red[tid + s]);
        __syncthreads();
    }
    float mx = red[0];
    __syncthreads();
    // masked exp + sum (softmax + mask renorm fold into one normalization)
    float w0 = mask[b * TT + tid]       * expf(e0 - mx);
    float w1 = mask[b * TT + tid + 256] * expf(e1 - mx);
    sm[tid] = w0; sm[tid + 256] = w1;
    red[tid] = w0 + w1;
    __syncthreads();
    for (int s = 128; s > 0; s >>= 1) {
        if (tid < s) red[tid] += red[tid + s];
        __syncthreads();
    }
    float inv = 1.0f / red[0];
    __syncthreads();
    sm[tid]       *= inv;
    sm[tid + 256] *= inv;
    __syncthreads();

    // ctx[b][h] = sum_t m[t] * values[t][b][h]
    int h    = tid & 127;
    int half = tid >> 7;
    float acc = 0.f;
    for (int t2 = half * 256; t2 < (half + 1) * 256; ++t2)
        acc += sm[t2] * values[((size_t)t2 * BB + b) * HH + h];
    red[tid] = acc;
    __syncthreads();
    if (tid < HH) {
        float cv = red[tid] + red[tid + 128];
        ctx[b * HH + tid] = cv;
        A_bf[b * K2 + HH + tid] = f2bf(cv);
        A_bf[b * K2 + tid]      = f2bf(sh2[tid]);
    }
}

// ---------------- per-step: pred = A(64x256) @ WpBF^T(256x32000) + bp, via bf16 WMMA ----------------
// Block: 256 threads = 8 waves. Each wave owns TWO 16-wide N tiles (full M=64) so that the
// 4 A fragments loaded per K-chunk are reused by 8 WMMAs (2 ds_load_b128 -> 2 WMMAs).
// Grid: 2000 N tiles / (8 waves * 2 tiles) = 125 blocks.
__global__ void __launch_bounds__(256)
pred_gemm(const __bf16* __restrict__ WpBF, const __bf16* __restrict__ A_bf,
          const float* __restrict__ bp, float* __restrict__ out, int t) {
    __shared__ __attribute__((aligned(32))) __bf16 sA[BB * K2];   // 32 KB

    // cooperative A load (64x256 bf16) into LDS
    {
        const uint4* src = (const uint4*)A_bf;
        uint4*       dst = (uint4*)sA;
        for (int i = threadIdx.x; i < BB * K2 / 8; i += 256) dst[i] = src[i];
    }
    __syncthreads();

    int wave   = threadIdx.x >> 5;
    int lane   = threadIdx.x & 31;
    int ntile0 = (blockIdx.x * 8 + wave) * 2;     // two adjacent N tiles per wave
    int nbase0 = ntile0 * 16;
    int nbase1 = nbase0 + 16;
    int ncol   = lane & 15;
    int khalf  = lane >> 4;

    v8f acc[2][4];
#pragma unroll
    for (int nt = 0; nt < 2; ++nt)
#pragma unroll
        for (int i = 0; i < 4; ++i)
#pragma unroll
            for (int r = 0; r < 8; ++r) acc[nt][i][r] = 0.0f;

    union AV { v16bf v; uint4 q[2]; };

#pragma unroll
    for (int kc = 0; kc < 8; ++kc) {
        // B fragments (32x16, K x N): lane ncol = column, khalf selects K 0-15 / 16-31.
        // B[k][n] = Wp[n][kc*32 + 16*khalf + k'] -> 16 contiguous bf16 (32B).
        v16bf bfrag0 = *(const v16bf*)(WpBF + (size_t)(nbase0 + ncol) * K2 + kc * 32 + 16 * khalf);
        v16bf bfrag1 = *(const v16bf*)(WpBF + (size_t)(nbase1 + ncol) * K2 + kc * 32 + 16 * khalf);

        // A fragments for all 4 M tiles, loaded once per K-chunk and reused by both N tiles.
        AV af[4];
#pragma unroll
        for (int mt = 0; mt < 4; ++mt) {
            // A (16x32): lane m = mt*16+ncol; element s -> K = 16*khalf + s
            // (16 contiguous elements, matching B).
            const __bf16* pa = sA + (size_t)(mt * 16 + ncol) * K2 + kc * 32 + 16 * khalf;
            af[mt].q[0] = *(const uint4*)(pa);
            af[mt].q[1] = *(const uint4*)(pa + 8);
        }

#pragma unroll
        for (int mt = 0; mt < 4; ++mt) {
            acc[0][mt] = __builtin_amdgcn_wmma_f32_16x16x32_bf16(
                false, af[mt].v, false, bfrag0, (short)0, acc[0][mt], false, false);
            acc[1][mt] = __builtin_amdgcn_wmma_f32_16x16x32_bf16(
                false, af[mt].v, false, bfrag1, (short)0, acc[1][mt], false, false);
        }
    }

#pragma unroll
    for (int nt = 0; nt < 2; ++nt) {
        int v = (nt == 0 ? nbase0 : nbase1) + ncol;
        float bv = bp[v];
#pragma unroll
        for (int mt = 0; mt < 4; ++mt) {
#pragma unroll
            for (int r = 0; r < 8; ++r) {
                int m = mt * 16 + r + 8 * khalf;   // C/D layout: lanes 0-15 M=r, 16-31 M=r+8
                out[(size_t)m * MAXLEN * VV + (size_t)t * VV + v] = acc[nt][mt][r] + bv;
            }
        }
    }
}

// ---------------- host launch ----------------
extern "C" void kernel_launch(void* const* d_in, const int* in_sizes, int n_in,
                              void* d_out, int out_size, void* d_ws, size_t ws_size,
                              hipStream_t stream) {
    const float* key    = (const float*)d_in[0];
    const float* values = (const float*)d_in[1];
    const float* mask   = (const float*)d_in[2];
    const float* emb    = (const float*)d_in[3];
    const float* Wih1   = (const float*)d_in[4];
    const float* Whh1   = (const float*)d_in[5];
    const float* bih1   = (const float*)d_in[6];
    const float* bhh1   = (const float*)d_in[7];
    const float* Wih2   = (const float*)d_in[8];
    const float* Whh2   = (const float*)d_in[9];
    const float* bih2   = (const float*)d_in[10];
    const float* bhh2   = (const float*)d_in[11];
    const float* Wq     = (const float*)d_in[12];
    const float* bq     = (const float*)d_in[13];
    const float* Wp     = (const float*)d_in[14];
    const float* bp     = (const float*)d_in[15];
    float* out = (float*)d_out;

    char* ws = (char*)d_ws;
    const size_t wpBytes = (size_t)VV * K2 * sizeof(__bf16);      // 16,384,000
    __bf16* WpBF  = (__bf16*)ws;
    float*  states = (float*)(ws + wpBytes);                      // h1,c1,h2,c2,ctx
    float*  h1  = states;
    float*  c1  = states + 1 * BB * HH;
    float*  h2  = states + 2 * BB * HH;
    float*  c2  = states + 3 * BB * HH;
    float*  ctx = states + 4 * BB * HH;
    float*  embed = states + 5 * BB * HH;
    __bf16* A_bf = (__bf16*)(ws + wpBytes + (size_t)6 * BB * HH * sizeof(float));

    // one-time: Wp -> bf16, zero recurrent state (ctx0 = h = c = 0)
    cvt_wp_bf16<<<4096, 256, 0, stream>>>(Wp, WpBF, VV * K2);
    zero_f32<<<(5 * BB * HH + 255) / 256, 256, 0, stream>>>(states, 5 * BB * HH);

    for (int t = 0; t < MAXLEN; ++t) {
        argmax_gather<<<BB, 256, 0, stream>>>(out, emb, embed, t);
        lstm_cell1<<<BB / 2, 256, 0, stream>>>(embed, ctx, Wih1, Whh1, bih1, bhh1, h1, c1);
        lstm_cell2<<<BB / 2, 256, 0, stream>>>(h1, Wih2, Whh2, bih2, bhh2, h2, c2);
        attention<<<BB, 256, 0, stream>>>(h2, key, values, mask, Wq, bq, ctx, A_bf);
        pred_gemm<<<(VV / 16) / 16, 256, 0, stream>>>(WpBF, A_bf, bp, out, t);
    }
}
